// SFM_43258910605610
// MI455X (gfx1250) — matrix operand order repaired and checked by reading, not compiled, not run on hardware
//
#include <hip/hip_runtime.h>

typedef _Float16 v16h __attribute__((ext_vector_type(16)));
typedef float    v8f  __attribute__((ext_vector_type(8)));

#define KNEI 8
#define DTC  0.02f
#define NEI_STRIDE 50   // floats per agent in LDS stage (8 rows * 6 + 2 pad)

// ---- CDNA5 async global->LDS copies (ASYNCcnt-tracked) ----
__device__ __forceinline__ void async_g2l_b64(const float* g, float* l) {
  unsigned lo = (unsigned)(uintptr_t)l;   // low 32 bits of generic LDS ptr = LDS byte offset
  asm volatile("global_load_async_to_lds_b64 %0, %1, off"
               :: "v"(lo), "v"(g) : "memory");
}
__device__ __forceinline__ void async_g2l_b32(const float* g, float* l) {
  unsigned lo = (unsigned)(uintptr_t)l;
  asm volatile("global_load_async_to_lds_b32 %0, %1, off"
               :: "v"(lo), "v"(g) : "memory");
}
__device__ __forceinline__ void wait_async_zero() {
  asm volatile("s_wait_asynccnt 0x0" ::: "memory");
}

// Build B matrix (32x16 f16) holding w2 broadcast along N:
// elements 0..7 carry K=8h+e (= w2[8h+e]), elements 8..15 carry K=16..31 (= 0).
__device__ __forceinline__ v16h make_w2_B(const float* __restrict__ w2, bool lo16) {
  union { _Float16 f[16]; v16h v; } B;
#pragma unroll
  for (int e = 0; e < 16; ++e)
    B.f[e] = (e < 8) ? (_Float16)w2[lo16 ? e : 8 + e] : (_Float16)0.0f;
  return B.v;
}

// Wave-cooperative mono MLP core: returns sum_e w2[e]*exp(-(w1[e]*x+b1[e]))
// for each lane's scalar x, using two V_WMMA_F32_16X16X32_F16 per 32 lanes.
__device__ __forceinline__ float wave_mono(float x, const float* __restrict__ w1,
                                           const float* __restrict__ b1,
                                           v16h B, int lane) {
  unsigned p[8];
#pragma unroll
  for (int j = 0; j < 8; ++j) {
    float h0 = __expf(-(w1[2 * j]     * x + b1[2 * j]));
    float h1 = __expf(-(w1[2 * j + 1] * x + b1[2 * j + 1]));
    union { _Float16 h[2]; unsigned u; } pk;
    pk.h[0] = (_Float16)h0;
    pk.h[1] = (_Float16)h1;
    p[j] = pk.u;
  }
  // Row->lane mapping:
  //  WMMA#1 rows 0..7 <- lanes 0..7,  rows 8..15 <- lanes 16..23
  //  WMMA#2 rows 0..7 <- lanes 8..15, rows 8..15 <- lanes 24..31
  const int  m      = lane & 15;
  const int  src1   = (m < 8) ? m : m + 8;
  const int  src2   = src1 + 8;
  const bool hihalf = lane >= 16;  // A lanes 16-31 carry K=8..15 of their row

  union { unsigned u[8]; v16h v; } A1, A2;
#pragma unroll
  for (int j = 0; j < 4; ++j) {
    unsigned lo1 = (unsigned)__shfl((int)p[j],     src1, 32);
    unsigned hi1 = (unsigned)__shfl((int)p[j + 4], src1, 32);
    unsigned lo2 = (unsigned)__shfl((int)p[j],     src2, 32);
    unsigned hi2 = (unsigned)__shfl((int)p[j + 4], src2, 32);
    A1.u[j] = hihalf ? hi1 : lo1;
    A2.u[j] = hihalf ? hi2 : lo2;
    A1.u[j + 4] = 0u;   // K = 16..31 unused (B is zero there)
    A2.u[j + 4] = 0u;
  }
  v8f cz = {};
  v8f d1 = __builtin_amdgcn_wmma_f32_16x16x32_f16(false, A1.v, false, B,
                                                  (short)0, cz, false, false);
  v8f d2 = __builtin_amdgcn_wmma_f32_16x16x32_f16(false, A2.v, false, B,
                                                  (short)0, cz, false, false);
  const bool hi8 = (lane & 8) != 0;
  const int  r   = lane & 7;
  float val = hi8 ? d2[0] : d1[0];
#pragma unroll
  for (int j = 1; j < 8; ++j) {
    float cj = hi8 ? d2[j] : d1[j];
    val = (r == j) ? cj : val;
  }
  return val;
}

__device__ __forceinline__ void angle_clamp_acc(float fx, float fy, float vx, float vy,
                                                float speed, float ang,
                                                float& ax, float& ay) {
  float num = vx * fx + vy * fy;
  float den = fmaxf(speed * sqrtf(fx * fx + fy * fy), 1e-8f);
  bool keep = fabsf(num / den) > ang;
  ax += keep ? fx : 0.0f;
  ay += keep ? fy : 0.0f;
}

__global__ __launch_bounds__(256) void SFM_43258910605610_kernel(
    const float* __restrict__ ego,    const float* __restrict__ nei,
    const float* __restrict__ border, const float* __restrict__ rt,
    const float* __restrict__ p_dest, const float* __restrict__ angle,
    const float* __restrict__ rep_w1, const float* __restrict__ rep_b1,
    const float* __restrict__ rep_w2, const float* __restrict__ rep_b2,
    const float* __restrict__ att_w1, const float* __restrict__ att_b1,
    const float* __restrict__ att_w2, const float* __restrict__ att_b2,
    const float* __restrict__ bor_w1, const float* __restrict__ bor_b1,
    const float* __restrict__ bor_w2, const float* __restrict__ bor_b2,
    const float* __restrict__ del_w1, const float* __restrict__ del_b1,
    const float* __restrict__ del_w2, const float* __restrict__ del_b2,
    float* __restrict__ out, int n_total)
{
  __shared__ float s_w1[4][16];  // 0=rep 1=att 2=bor 3=del
  __shared__ float s_b1[4][16];
  __shared__ float s_nei[256 * NEI_STRIDE];  // per agent: 8 rows x (id,px,py,vx,vy,pad)

  const int t    = threadIdx.x;
  const int base = blockIdx.x * 256;

  // ---- async-stage neighbor columns 0..4 of this block's 2048 rows into LDS ----
  // row slot: 6 floats, 8B-aligned (stride 50 floats per agent; 50*4 % 8 == 0)
#pragma unroll
  for (int j = 0; j < 8; ++j) {
    const int rid = j * 256 + t;            // 0..2047, consecutive across lanes
    const int a   = rid >> 3;               // agent within block
    const int k   = rid & 7;                // neighbor row
    int ga = base + a;
    if (ga >= n_total) ga = n_total - 1;    // duplicate-stage tail (distinct dst slots)
    const float* src = nei + ((size_t)ga * KNEI + k) * 16;
    float*       dst = &s_nei[a * NEI_STRIDE + k * 6];
    async_g2l_b64(src,     dst);        // id, posx
    async_g2l_b64(src + 2, dst + 2);    // posy, velx
    async_g2l_b32(src + 4, dst + 4);    // vely
  }

  if (t < 16)      {                 s_w1[0][t] = rep_w1[t]; s_b1[0][t] = rep_b1[t]; }
  else if (t < 32) { int e = t - 16; s_w1[1][e] = att_w1[e]; s_b1[1][e] = att_b1[e]; }
  else if (t < 48) { int e = t - 32; s_w1[2][e] = bor_w1[e]; s_b1[2][e] = bor_b1[e]; }
  else if (t < 64) { int e = t - 48; s_w1[3][e] = del_w1[e]; s_b1[3][e] = del_b1[e]; }

  wait_async_zero();     // our wave's async copies landed in LDS
  __syncthreads();       // everyone's copies visible

  const int  lane = t & 31;
  const bool lo16 = lane < 16;
  const v16h Brep = make_w2_B(rep_w2, lo16);
  const v16h Batt = make_w2_B(att_w2, lo16);
  const v16h Bbor = make_w2_B(bor_w2, lo16);
  const v16h Bdel = make_w2_B(del_w2, lo16);
  const float rep_b2s = rep_b2[0], att_b2s = att_b2[0];
  const float bor_b2s = bor_b2[0], del_b2s = del_b2[0];

  int n = base + t;
  if (n >= n_total) n = n_total - 1;  // keep EXEC full for WMMA; tail duplicates work

  __builtin_prefetch(rt + (size_t)n * (KNEI * 2), 0, 1);

  // ---- ego row (cols: 1,2 pos; 3,4 vel; 7..14 ids) ----
  const float4* egoR = reinterpret_cast<const float4*>(ego + (size_t)n * 16);
  const float4 e0 = egoR[0], e1 = egoR[1], e2v = egoR[2], e3v = egoR[3];
  const float ex = e0.y, ey = e0.z;
  const float vx = e0.w, vy = e1.x;
  const float egoid[8] = { e1.w, e2v.x, e2v.y, e2v.z, e2v.w, e3v.x, e3v.y, e3v.z };
  const float speed = sqrtf(vx * vx + vy * vy);
  const float ang = angle[0];
  const float pd0 = p_dest[0], pd1 = p_dest[1];

  const float* myNei = &s_nei[t * NEI_STRIDE];

  // ---- neighbor ids / recording buffer ----
  float nid[KNEI], bid[KNEI], bct[KNEI];
#pragma unroll
  for (int k = 0; k < KNEI; ++k) {
    nid[k] = myNei[k * 6];
    const float2 rr = *reinterpret_cast<const float2*>(rt + ((size_t)n * KNEI + k) * 2);
    bid[k] = rr.x; bct[k] = rr.y;
  }
  bool idxm[KNEI], ifin[KNEI];
#pragma unroll
  for (int k = 0; k < KNEI; ++k) {
    bool inEgo = false, bInN = false;
#pragma unroll
    for (int j = 0; j < KNEI; ++j)
      bInN = bInN || (bid[k] == nid[j]);
#pragma unroll
    for (int j = 0; j < 8; ++j) inEgo = inEgo || (nid[k] == egoid[j]);
    idxm[k] = inEgo && (nid[k] != 0.0f);
    ifin[k] = bInN;
  }
  // Sorted-candidate count: position p gets (bct+1) of the p-th ifin slot, else 1.0
  // ("new" entries and the fill value both contribute 1.0)
  float count[KNEI];
#pragma unroll
  for (int p = 0; p < KNEI; ++p) {
    float cv = 1.0f; int seen = 0;
#pragma unroll
    for (int k = 0; k < KNEI; ++k) {
      bool hit = ifin[k] && (seen == p);
      cv = hit ? (bct[k] + 1.0f) : cv;
      seen += ifin[k] ? 1 : 0;
    }
    count[p] = cv;
  }

  // ---- neighbor forces ----
  float fnx = 0.0f, fny = 0.0f;
#pragma unroll
  for (int k = 0; k < KNEI; ++k) {
    const float* nb = myNei + k * 6;
    const float px = nb[1], py = nb[2], nvxk = nb[3], nvyk = nb[4];
    const bool idx = idxm[k];

    const float r0 = idx ? (px - ex) : 0.0f;
    const float r1 = idx ? (py - ey) : 0.0f;
    const float rn = sqrtf(r0 * r0 + r1 * r1);
    const float rns = idx ? rn : 1.0f;
    const float d0 = r0 / rns, d1 = r1 / rns;

    const float mdel = wave_mono(count[k], s_w1[3], s_b1[3], Bdel, lane) + del_b2s;
    const float matt = wave_mono(rn,       s_w1[1], s_b1[1], Batt, lane) + att_b2s;
    const float fa = mdel * matt;

    const float t0 = r0 + nvxk * DTC, t1 = r1 + nvyk * DTC;
    const float bsq = rn + t0 * t0 + t1 * t1
                    - (nvxk * DTC) * (nvxk * DTC) - (nvyk * DTC) * (nvyk * DTC);
    const float bb = sqrtf(fmaxf(idx ? bsq : 1.0f, 1e-12f)) * 0.5f;
    const float mrep = wave_mono(bb, s_w1[0], s_b1[0], Brep, lane) + rep_b2s;

    const float fax = idx ? fa * d0 : 0.0f,   fay = idx ? fa * d1 : 0.0f;
    const float frx = idx ? mrep * d0 : 0.0f, fry = idx ? mrep * d1 : 0.0f;
    angle_clamp_acc(fax, fay, vx, vy, speed, ang, fnx, fny);
    angle_clamp_acc(frx, fry, vx, vy, speed, ang, fnx, fny);
  }

  // ---- destination force ----
  float fdax = 0.0f, fday = 0.0f;
  {
    const float fdx = (pd1 * speed - vx) / pd0;
    const float fdy = (-vy) / pd0;
    angle_clamp_acc(fdx, fdy, vx, vy, speed, ang, fdax, fday);
  }

  // ---- border force (ego col 2 vs border[0], border[3]) ----
  float fbx = 0.0f, fby = 0.0f;
  {
    const float rb0 = ey - border[0];
    const float rb1 = ey - border[3];
    const float rbn0 = fabsf(rb0), rbn1 = fabsf(rb1);
    const float m0 = wave_mono(rbn0, s_w1[2], s_b1[2], Bbor, lane) + bor_b2s;
    const float m1 = wave_mono(rbn1, s_w1[2], s_b1[2], Bbor, lane) + bor_b2s;
    const float fy0 = m0 * (rb0 / rbn0);
    const float fy1 = m1 * (rb1 / rbn1);
    angle_clamp_acc(0.0f, fy0, vx, vy, speed, ang, fbx, fby);
    angle_clamp_acc(0.0f, fy1, vx, vy, speed, ang, fbx, fby);
  }

  float* o = out + (size_t)n * 6;
  o[0] = fdax; o[1] = fday;
  o[2] = fnx;  o[3] = fny;
  o[4] = fbx;  o[5] = fby;
}

extern "C" void kernel_launch(void* const* d_in, const int* in_sizes, int n_in,
                              void* d_out, int out_size, void* d_ws, size_t ws_size,
                              hipStream_t stream) {
  (void)n_in; (void)out_size; (void)d_ws; (void)ws_size;
  const float* ego    = (const float*)d_in[0];
  const float* nei    = (const float*)d_in[1];
  const float* border = (const float*)d_in[2];
  const float* rt     = (const float*)d_in[3];
  const float* p_dest = (const float*)d_in[4];
  const float* angle  = (const float*)d_in[5];
  const float* rep_w1 = (const float*)d_in[6];
  const float* rep_b1 = (const float*)d_in[7];
  const float* rep_w2 = (const float*)d_in[8];
  const float* rep_b2 = (const float*)d_in[9];
  const float* att_w1 = (const float*)d_in[10];
  const float* att_b1 = (const float*)d_in[11];
  const float* att_w2 = (const float*)d_in[12];
  const float* att_b2 = (const float*)d_in[13];
  const float* bor_w1 = (const float*)d_in[14];
  const float* bor_b1 = (const float*)d_in[15];
  const float* bor_w2 = (const float*)d_in[16];
  const float* bor_b2 = (const float*)d_in[17];
  const float* del_w1 = (const float*)d_in[18];
  const float* del_b1 = (const float*)d_in[19];
  const float* del_w2 = (const float*)d_in[20];
  const float* del_b2 = (const float*)d_in[21];
  float* out = (float*)d_out;

  const int n_total = in_sizes[0] / 16;
  const int block = 256;
  const int grid = (n_total + block - 1) / block;
  SFM_43258910605610_kernel<<<grid, block, 0, stream>>>(
      ego, nei, border, rt, p_dest, angle,
      rep_w1, rep_b1, rep_w2, rep_b2,
      att_w1, att_b1, att_w2, att_b2,
      bor_w1, bor_b1, bor_w2, bor_b2,
      del_w1, del_b1, del_w2, del_b2,
      out, n_total);
}
